// DeformableCrossTemporalAttention_3186865733843
// MI455X (gfx1250) — matrix operand order repaired and checked by reading, not compiled, not run on hardware
//
#include <hip/hip_runtime.h>
#include <hip/hip_bf16.h>
#include <math.h>

typedef _Float16 f16;
typedef __attribute__((ext_vector_type(4)))  _Float16 v4h;
typedef __attribute__((ext_vector_type(8)))  _Float16 v8h;
typedef __attribute__((ext_vector_type(16))) _Float16 v16h;
typedef __attribute__((ext_vector_type(8)))  float    v8f;

typedef unsigned int u32;
typedef __attribute__((ext_vector_type(4))) u32 u32x4;
typedef __attribute__((ext_vector_type(4))) int i32x4;
typedef __attribute__((ext_vector_type(8))) int i32x8;

#if defined(__has_builtin)
# if __has_builtin(__builtin_amdgcn_tensor_load_to_lds)
#  define HAVE_TDM 1
#  warning "CDNA5 probe: __builtin_amdgcn_tensor_load_to_lds AVAILABLE (TDM path compiled)"
# else
#  warning "CDNA5 probe: tensor_load_to_lds builtin NOT available (fallback copy path)"
# endif
#endif

// Problem constants (from reference): B=2, C=256, H=W=64, T=7, nH=8, K=4, hd=32
#define HW    4096
#define CDIM  256

// ---------------------------------------------------------------------------
// Layout-conversion kernels (f32 NCHW -> f16 NHWC, weight permutes)
// ---------------------------------------------------------------------------
__global__ void cvt_nchw_nhwc(const float* __restrict__ src, f16* __restrict__ dst,
                              int C, int hw, long total) {
  long i = (long)blockIdx.x * blockDim.x + threadIdx.x;
  if (i >= total) return;
  int c  = (int)(i % C);
  long r = i / C;
  int yx = (int)(r % hw);
  long n = r / hw;
  dst[i] = (f16)src[(n * C + c) * hw + yx];
}

// (O,I,3,3) f32 -> (9, O, I) f16  so B-fragment K-runs are contiguous
__global__ void perm_w3x3(const float* __restrict__ src, f16* __restrict__ dst,
                          int O, int I, long total) {
  long i = (long)blockIdx.x * blockDim.x + threadIdx.x;
  if (i >= total) return;
  int ii  = (int)(i % I);
  long r  = i / I;
  int oc  = (int)(r % O);
  int tap = (int)(r / O);
  dst[i] = (f16)src[((long)oc * I + ii) * 9 + tap];
}

__global__ void cvt_f32_f16(const float* __restrict__ src, f16* __restrict__ dst, long total) {
  long i = (long)blockIdx.x * blockDim.x + threadIdx.x;
  if (i >= total) return;
  dst[i] = (f16)src[i];
}

// ---------------------------------------------------------------------------
// 3x3 conv (pad=1) + exact GELU, implicit GEMM via v_wmma_f32_16x16x32_f16.
// The whole haloed input slab (3 rows x 18 px x 256 ch f16 = 27KB, zero-padded
// at borders) is staged in LDS ONCE, then all 9 taps x 8 k-blocks of WMMA run
// against read-only LDS with a single barrier (no per-iteration sync).
// grid = (B*H*(W/16), Cout/64), block = 128 (4 waves)
// ---------------------------------------------------------------------------
__global__ __launch_bounds__(128)
void conv3x3_gelu_wmma(const f16* __restrict__ X,    // (B,H,W,C) f16
                       const f16* __restrict__ Wt,   // (9, Cout, Cin) f16
                       const float* __restrict__ bias,
                       f16* __restrict__ Y)          // (B*H*W, Cout) f16
{
  __shared__ __align__(16) _Float16 sS[3 * 18 * 256];   // [row 0..2][px 0..17][ch]
  const int tid  = threadIdx.x;
  const int lane = tid & 31;
  const int wid  = tid >> 5;
  const int mt   = blockIdx.x;
  const int x0   = (mt & 3) << 4;       // W/16 = 4 tiles per row
  const int y    = (mt >> 2) & 63;
  const int b    = mt >> 8;
  const int ntile = blockIdx.y * 4 + wid;
  const int n16  = lane & 15;
  const int hi16 = lane >> 4;
  const int kkA  = hi16 * 8;            // unused; A frag uses kkB: 16 contiguous K per lane
  const int kkB  = hi16 * 16;           // B frag: 16 contiguous K per lane
  const int mrow = lane & 15;

  // ---- stage haloed slab: 3*18*256 halfs = 1728 x v8h chunks, zero outside
  for (int c = tid; c < 1728; c += 128) {
    const int e  = c * 8;
    const int ch = e & 255;
    const int pr = e >> 8;              // 0..53
    const int px = pr % 18;
    const int ry = pr / 18;
    const int yi = y + ry - 1;
    const int xi = x0 + px - 1;
    v8h fill = {};
    if (yi >= 0 && yi < 64 && xi >= 0 && xi < 64)
      fill = *(const v8h*)(X + (((size_t)(b * 64 + yi) * 64 + xi) << 8) + ch);
    *(v8h*)(&sS[e]) = fill;
  }
  __syncthreads();

  v8f acc = {};
  for (int tap = 0; tap < 9; ++tap) {
    const int dy = tap / 3, dx = tap % 3;            // slab-relative (0..2)
    const f16* rp = &sS[((dy * 18) + (mrow + dx)) * 256];
    const f16* wrow = Wt + ((size_t)(tap * CDIM + ntile * 16 + n16) << 8);
    #pragma unroll
    for (int kb = 0; kb < 8; ++kb) {
      const int k0 = kb * 32;
      v16h a;
      { // A fragment: 16 contiguous K per lane, same order as B (two ds_load_b128)
        v8h lo = *(const v8h*)(rp + k0 + kkB);
        v8h hh = *(const v8h*)(rp + k0 + 8 + kkB);
        #pragma unroll
        for (int i = 0; i < 8; ++i) { a[i] = lo[i]; a[i + 8] = hh[i]; }
      }
      // B fragment: column n16, 16 contiguous K (32B aligned)
      v16h bf = *(const v16h*)(wrow + k0 + kkB);
      acc = __builtin_amdgcn_wmma_f32_16x16x32_f16(false, a, false, bf, (short)0, acc,
                                                   false, false);
    }
  }
  #pragma unroll
  for (int r = 0; r < 8; ++r) {   // C/D layout: M = r + 8*hi16, N = n16
    const int m = r + hi16 * 8;
    const int n = ntile * 16 + n16;
    float v = acc[r] + bias[n];
    float g = 0.5f * v * (1.f + erff(v * 0.70710678118654752f));   // exact GELU
    Y[(((size_t)(b * 64 + y) * 64 + (x0 + m)) << 8) + n] = (f16)g;
  }
}

// ---------------------------------------------------------------------------
// 1x1 conv GEMM via WMMA with fused epilogues.  The 16x256 f16 A panel (8KB)
// is staged in LDS once per block via the Tensor Data Mover (one 2D-tile D#,
// issued by wave 0, TENSORcnt-waited, then barrier) and reused by 4 waves.
//   mode 0: offsets  -> tanh(v)*32, f32 (B,448,HW) channel-major
//   mode 1: logits   -> f32 (B,224,HW) channel-major
//   mode 2: value    -> f16 scatter to (B, head, t, yx, hd)
//   mode 3: final    -> f32 (B,256,HW) + residual (query)
// grid = (P/16, ceil(Cout/64)), block = 128
// ---------------------------------------------------------------------------
__global__ __launch_bounds__(128)
void conv1x1_wmma(const f16* __restrict__ X,   // (P, 256) f16
                  const f16* __restrict__ Wt,  // (Cout, 256) f16
                  const float* __restrict__ bias,
                  int Cout, int mode,
                  float* __restrict__ outF,
                  f16* __restrict__ outH,
                  const float* __restrict__ resid)
{
  __shared__ __align__(16) _Float16 sX[16 * 256];  // A panel: 16 rows x 256 ch
  const int tid = threadIdx.x, lane = tid & 31, wid = tid >> 5;
  const int p0  = blockIdx.x * 16;

#ifdef HAVE_TDM
  if (tid == 0) {
    // Tensor DMA Descriptor (D#): 2D tile, 2-byte elements, 256 x 16,
    // row stride 256 elements; dest = sX.  (CDNA5 ISA)
    const unsigned lds_off = (unsigned)(size_t)(&sX[0]);       // addr[31:0] = LDS offset
    const unsigned long long ga = (unsigned long long)(size_t)(X + ((size_t)p0 << 8));
    u32x4 g0;
    g0[0] = 1u;                                        // count=1, user descriptor
    g0[1] = lds_off;                                   // lds_addr
    g0[2] = (u32)(ga & 0xFFFFFFFFu);                   // global_addr[31:0]
    g0[3] = (u32)((ga >> 32) & 0x1FFFFFFu) | (2u << 30); // global_addr[56:32] | type=2
    i32x8 g1;
    g1[0] = (int)(1u << 16);                 // workgroup_mask=0, data_size=1 (2B)
    g1[1] = (int)(256u << 16);               // tensor_dim0[15:0]=256 @ bits 63:48
    g1[2] = (int)(16u << 16);                // dim0 hi=0 | tensor_dim1[15:0]=16
    g1[3] = (int)(256u << 16);               // dim1 hi=0 | tile_dim0=256
    g1[4] = (int)16;                         // tile_dim1=16, tile_dim2=0
    g1[5] = (int)256;                        // tensor_dim0_stride[31:0]=256
    g1[6] = 0;                               // stride hi / dim1_stride lo
    g1[7] = 0;
    i32x4 gz = {0, 0, 0, 0};                 // groups 2/3 unused (<=2D tensor)
#if __clang_major__ >= 23
    i32x8 gz8 = {0, 0, 0, 0, 0, 0, 0, 0};
    __builtin_amdgcn_tensor_load_to_lds(g0, g1, gz, gz, gz8, 0);
#else
    __builtin_amdgcn_tensor_load_to_lds(g0, g1, gz, gz, 0);
#endif
    __builtin_amdgcn_s_wait_tensorcnt(0);
  }
#else
  { // fallback: cooperative vector copy (16B per thread x 4 iters)
    const f16* src = X + ((size_t)p0 << 8);
    #pragma unroll
    for (int it = 0; it < 4; ++it) {
      int e = (it * 128 + tid) * 8;
      *(v8h*)(&sX[e]) = *(const v8h*)(src + e);
    }
  }
#endif
  __syncthreads();

  const int ntile = blockIdx.y * 4 + wid;
  if (ntile * 16 >= Cout) return;             // wave-uniform guard (EXEC stays full)
  const int n16  = lane & 15;
  const int hi16 = lane >> 4;
  const int kkA  = hi16 * 8, kkB = hi16 * 16, mrow = lane & 15;
  const f16* xr = &sX[mrow * 256];
  const f16* wr = Wt + ((size_t)(ntile * 16 + n16) << 8);

  v8f acc = {};
  #pragma unroll
  for (int kb = 0; kb < 8; ++kb) {
    const int k0 = kb * 32;
    v16h a;
    { v8h lo = *(const v8h*)(xr + k0 + kkB);       // ds_load_b128
      v8h hh = *(const v8h*)(xr + k0 + 8 + kkB);
      #pragma unroll
      for (int i = 0; i < 8; ++i) { a[i] = lo[i]; a[i + 8] = hh[i]; } }
    v16h bf = *(const v16h*)(wr + k0 + kkB);
    acc = __builtin_amdgcn_wmma_f32_16x16x32_f16(false, a, false, bf, (short)0, acc,
                                                 false, false);
  }
  #pragma unroll
  for (int r = 0; r < 8; ++r) {
    const int m = r + hi16 * 8;
    const int p = p0 + m;
    const int n = ntile * 16 + n16;
    float v = acc[r] + bias[n];
    if (mode == 0) {                 // offsets: tanh clamp to +/-32 px
      int b = p >> 12, yx = p & 4095;
      outF[((size_t)(b * 448 + n) << 12) + yx] = tanhf(v) * 32.f;
    } else if (mode == 1) {          // attention logits
      int b = p >> 12, yx = p & 4095;
      outF[((size_t)(b * 224 + n) << 12) + yx] = v;
    } else if (mode == 2) {          // value: P rows are (t*B+b)*HW + yx
      int tb = p >> 12, yx = p & 4095;
      int t = tb >> 1, b = tb & 1;
      int head = n >> 5, d = n & 31;
      outH[(((((size_t)(b * 8 + head) * 7 + t) << 12) + yx) << 5) + d] = (f16)v;
    } else {                         // final 1x1 + residual -> d_out (f32 NCHW)
      int b = p >> 12, yx = p & 4095;
      size_t o = ((size_t)(b * 256 + n) << 12) + yx;
      outF[o] = resid[o] + v;
    }
  }
}

// ---------------------------------------------------------------------------
// softmax over the 28 samples per (b, head, pixel); in-place on (B,224,HW)
// ---------------------------------------------------------------------------
__global__ void softmax28(float* __restrict__ attn) {
  int idx = blockIdx.x * blockDim.x + threadIdx.x;   // 2*8*4096
  int b  = idx >> 15;
  int h  = (idx >> 12) & 7;
  int yx = idx & 4095;
  float* base = attn + (((size_t)(b * 224 + h * 28)) << 12) + yx;
  float l[28];
  float mx = -1e30f;
  #pragma unroll
  for (int j = 0; j < 28; ++j) { l[j] = base[(size_t)j << 12]; mx = fmaxf(mx, l[j]); }
  float s = 0.f;
  #pragma unroll
  for (int j = 0; j < 28; ++j) { l[j] = __expf(l[j] - mx); s += l[j]; }
  float inv = 1.f / s;
  #pragma unroll
  for (int j = 0; j < 28; ++j) base[(size_t)j << 12] = l[j] * inv;
}

// ---------------------------------------------------------------------------
// Deformable bilinear sampling + attention aggregation.
// One wave per (b, pixel, head); lane = hd element -> every corner gather is
// one coalesced 64B wave read from the f16 value tensor.
// agg rows: (b*HW + yx, head*32 + d)  f16, feeds the final 1x1 WMMA conv.
// ---------------------------------------------------------------------------
__global__ __launch_bounds__(256)
void sample_agg(const float* __restrict__ offs, const float* __restrict__ attn,
                const f16* __restrict__ vbuf, f16* __restrict__ agg)
{
  const int lane = threadIdx.x & 31;
  const int wg   = blockIdx.x * 8 + (threadIdx.x >> 5);  // 2*4096*8 waves
  const int b    = wg >> 15;
  const int rem  = wg & 32767;
  const int yx   = rem >> 3;
  const int h    = rem & 7;
  const int x = yx & 63, y = yx >> 6;

  float acc = 0.f;
  for (int j = 0; j < 28; ++j) {
    const int t = j >> 2, k = j & 3;
    const float aw = attn[(((size_t)(b * 224 + h * 28 + j)) << 12) + yx];
    const int och  = ((h * 7 + t) * 4 + k) * 2;
    const float ox = offs[(((size_t)(b * 448 + och)) << 12) + yx];
    const float oy = offs[(((size_t)(b * 448 + och + 1)) << 12) + yx];
    if (j < 27)  // warm GL2 for the next sample's attention weight
      __builtin_prefetch(&attn[(((size_t)(b * 224 + h * 28 + j + 1)) << 12) + yx], 0, 0);
    const float px = (float)x + ox, py = (float)y + oy;
    const float x0f = floorf(px), y0f = floorf(py);
    const int   x0 = (int)x0f,    y0 = (int)y0f;
    const float wx1 = px - x0f,   wy1 = py - y0f;
    const float wx0 = 1.f - wx1,  wy0 = 1.f - wy1;
    const size_t vb0 = ((size_t)((b * 8 + h) * 7 + t)) << 17;  // *HW*32

    auto corner = [&](int xi, int yi, float wgt) {
      if (xi >= 0 && xi < 64 && yi >= 0 && yi < 64) {
        float val = (float)vbuf[vb0 + (((size_t)(yi * 64 + xi)) << 5) + lane];
        acc += aw * wgt * val;
      }
    };
    corner(x0,     y0,     wx0 * wy0);
    corner(x0 + 1, y0,     wx1 * wy0);
    corner(x0,     y0 + 1, wx0 * wy1);
    corner(x0 + 1, y0 + 1, wx1 * wy1);
  }
  agg[(((size_t)((b << 12) + yx)) << 8) + h * 32 + lane] = (f16)acc;
}

// ---------------------------------------------------------------------------
extern "C" void kernel_launch(void* const* d_in, const int* in_sizes, int n_in,
                              void* d_out, int out_size, void* d_ws, size_t ws_size,
                              hipStream_t stream) {
  const float* q      = (const float*)d_in[0];   // (2,256,64,64)
  const float* ctx    = (const float*)d_in[1];   // (7,2,256,64,64)
  const float* w_off1 = (const float*)d_in[2];
  const float* b_off1 = (const float*)d_in[3];
  const float* w_off2 = (const float*)d_in[4];
  const float* b_off2 = (const float*)d_in[5];
  const float* w_at1  = (const float*)d_in[6];
  const float* b_at1  = (const float*)d_in[7];
  const float* w_at2  = (const float*)d_in[8];
  const float* b_at2  = (const float*)d_in[9];
  const float* w_v    = (const float*)d_in[10];
  const float* b_v    = (const float*)d_in[11];
  const float* w_out  = (const float*)d_in[12];
  const float* b_out  = (const float*)d_in[13];
  float* out = (float*)d_out;

  // workspace carve (256B aligned)
  char* ws = (char*)d_ws;
  size_t off = 0;
  auto carve = [&](size_t bytes) -> void* {
    off = (off + 255) & ~(size_t)255;
    void* p = ws + off;
    off += bytes;
    return p;
  };
  f16*   qx     = (f16*)carve((size_t)2 * HW * CDIM * 2);        // query NHWC f16
  f16*   ctxh   = (f16*)carve((size_t)14 * HW * CDIM * 2);       // context NHWC f16
  f16*   w1t    = (f16*)carve((size_t)9 * 256 * 256 * 2);        // off1 (tap,O,I)
  f16*   w2t    = (f16*)carve((size_t)9 * 256 * 256 * 2);        // at1
  f16*   woff2h = (f16*)carve((size_t)448 * 256 * 2);
  f16*   wat2h  = (f16*)carve((size_t)224 * 256 * 2);
  f16*   wvh    = (f16*)carve((size_t)256 * 256 * 2);
  f16*   wouth  = (f16*)carve((size_t)256 * 256 * 2);
  f16*   h1     = (f16*)carve((size_t)2 * HW * CDIM * 2);        // gelu(off1 conv)
  f16*   ha     = (f16*)carve((size_t)2 * HW * CDIM * 2);        // gelu(at1 conv)
  float* offsB  = (float*)carve((size_t)2 * 448 * HW * 4);
  float* attnB  = (float*)carve((size_t)2 * 224 * HW * 4);
  f16*   vbuf   = (f16*)carve((size_t)2 * 8 * 7 * HW * 32 * 2);  // (b,h,t,yx,hd)
  f16*   aggB   = (f16*)carve((size_t)2 * HW * CDIM * 2);

  // 1) precision / layout conversions
  {
    long tq = (long)2 * HW * CDIM;
    cvt_nchw_nhwc<<<(int)((tq + 255) / 256), 256, 0, stream>>>(q, qx, CDIM, HW, tq);
    long tc = (long)14 * HW * CDIM;
    cvt_nchw_nhwc<<<(int)((tc + 255) / 256), 256, 0, stream>>>(ctx, ctxh, CDIM, HW, tc);
    long t3 = 9L * 256 * 256;
    perm_w3x3<<<(int)((t3 + 255) / 256), 256, 0, stream>>>(w_off1, w1t, 256, 256, t3);
    perm_w3x3<<<(int)((t3 + 255) / 256), 256, 0, stream>>>(w_at1, w2t, 256, 256, t3);
    cvt_f32_f16<<<(448 * 256) / 256, 256, 0, stream>>>(w_off2, woff2h, 448L * 256);
    cvt_f32_f16<<<(224 * 256) / 256, 256, 0, stream>>>(w_at2, wat2h, 224L * 256);
    cvt_f32_f16<<<(256 * 256) / 256, 256, 0, stream>>>(w_v, wvh, 256L * 256);
    cvt_f32_f16<<<(256 * 256) / 256, 256, 0, stream>>>(w_out, wouth, 256L * 256);
  }

  // 2) 3x3 conv + GELU branches (WMMA implicit GEMM, single-stage LDS slab)
  conv3x3_gelu_wmma<<<dim3(512, 4), 128, 0, stream>>>(qx, w1t, b_off1, h1);
  conv3x3_gelu_wmma<<<dim3(512, 4), 128, 0, stream>>>(qx, w2t, b_at1, ha);

  // 3) offsets head (Cout=448), attn logits (Cout=224)
  conv1x1_wmma<<<dim3(512, 7), 128, 0, stream>>>(h1, woff2h, b_off2, 448, 0,
                                                 offsB, nullptr, nullptr);
  conv1x1_wmma<<<dim3(512, 4), 128, 0, stream>>>(ha, wat2h, b_at2, 224, 1,
                                                 attnB, nullptr, nullptr);
  softmax28<<<256, 256, 0, stream>>>(attnB);

  // 4) value projection on all T*B frames, scattered to sampling layout
  conv1x1_wmma<<<dim3(3584, 4), 128, 0, stream>>>(ctxh, wvh, b_v, 256, 2,
                                                  nullptr, vbuf, nullptr);

  // 5) deformable bilinear sampling + attention-weighted aggregation
  sample_agg<<<8192, 256, 0, stream>>>(offsB, attnB, vbuf, aggB);

  // 6) output projection + residual -> d_out (f32 NCHW)
  conv1x1_wmma<<<dim3(512, 4), 128, 0, stream>>>(aggB, wouth, b_out, 256, 3,
                                                 out, nullptr, q);
}
